// LogMMExpModel_32564442038608
// MI455X (gfx1250) — matrix operand order repaired and checked by reading, not compiled, not run on hardware
//
#include <hip/hip_runtime.h>
#include <hip/hip_bf16.h>

typedef __attribute__((ext_vector_type(16))) _Float16 v16h;
typedef __attribute__((ext_vector_type(8)))  _Float16 v8h;
typedef __attribute__((ext_vector_type(4)))  _Float16 v4h;
typedef __attribute__((ext_vector_type(8)))  float    v8f;
typedef unsigned int  u32x4 __attribute__((ext_vector_type(4)));
typedef int           i32x8 __attribute__((ext_vector_type(8)));
typedef int           i32x4 __attribute__((ext_vector_type(4)));

union V16U { v16h v; v8h h[2]; };

#define N_DIM 4096
#define D_DIM 1024
#define E_DIM 4096

// GEMM tiling
#define BLOCK_M 128
#define BLOCK_N 128
#define KC      64                 // K-chunk per LDS stage (halves)
#define ROWB    144                // LDS row stride: 128B data + 16B TDM pad
#define TILE_BYTES (128 * ROWB)    // one 128-row tile in LDS (18 KB)
#define BUF_BYTES  (2 * TILE_BYTES)

#if __has_builtin(__builtin_amdgcn_tensor_load_to_lds)
#define USE_TDM 1
#else
#define USE_TDM 0
#endif

// ---------------------------------------------------------------------------
// Kernel 1: per-row max of x, then ex[r, d] = (f16) exp(x[r,d] - xm[r])
// ---------------------------------------------------------------------------
__global__ __launch_bounds__(256) void rowmax_exp_kernel(
    const float* __restrict__ x, _Float16* __restrict__ ex,
    float* __restrict__ xm) {
  __shared__ float red[256];
  const int row = blockIdx.x;
  const int tid = threadIdx.x;
  const float4 v = *(const float4*)(x + (size_t)row * D_DIM + tid * 4);
  float m = fmaxf(fmaxf(v.x, v.y), fmaxf(v.z, v.w));
  red[tid] = m;
  __syncthreads();
  for (int s = 128; s > 0; s >>= 1) {
    if (tid < s) red[tid] = fmaxf(red[tid], red[tid + s]);
    __syncthreads();
  }
  m = red[0];
  if (tid == 0) xm[row] = m;
  v4h e;
  e[0] = (_Float16)__expf(v.x - m);
  e[1] = (_Float16)__expf(v.y - m);
  e[2] = (_Float16)__expf(v.z - m);
  e[3] = (_Float16)__expf(v.w - m);
  *(v4h*)(ex + (size_t)row * D_DIM + tid * 4) = e;
}

// ---------------------------------------------------------------------------
// Kernel 2: per-column max of A (D x E)
// ---------------------------------------------------------------------------
__global__ __launch_bounds__(256) void colmax_kernel(
    const float* __restrict__ A, float* __restrict__ am) {
  const int e = blockIdx.x * 256 + threadIdx.x;
  float m = -3.402823466e+38f;
  for (int d = 0; d < D_DIM; ++d)
    m = fmaxf(m, A[(size_t)d * E_DIM + e]);
  am[e] = m;
}

// ---------------------------------------------------------------------------
// Kernel 3: LDS-tiled transpose+exp: eaT[e, d] = (f16) exp(A[d, e] - am[e])
// ---------------------------------------------------------------------------
__global__ __launch_bounds__(256) void colexp_transpose_kernel(
    const float* __restrict__ A, const float* __restrict__ am,
    _Float16* __restrict__ eaT) {
  __shared__ float tile[32][33];
  const int e0 = blockIdx.x * 32;
  const int d0 = blockIdx.y * 32;
  const int tx = threadIdx.x;
  const int ty = threadIdx.y;
#pragma unroll
  for (int i = 0; i < 4; ++i) {
    const int dl = ty + i * 8;
    tile[dl][tx] = A[(size_t)(d0 + dl) * E_DIM + e0 + tx];
  }
  __syncthreads();
#pragma unroll
  for (int i = 0; i < 4; ++i) {
    const int el = ty + i * 8;
    const int e  = e0 + el;
    eaT[(size_t)e * D_DIM + d0 + tx] =
        (_Float16)__expf(tile[tx][el] - am[e]);
  }
}

// ---------------------------------------------------------------------------
// TDM descriptor issue: 2D tile, 128 rows x KC halves, from a row-major
// matrix with row stride D_DIM halves, into LDS with 16B pad per 128B row.
// D# packing per CDNA5 ISA 8.3/8.4.  6-arg builtin form (clang-23 lane).
// ---------------------------------------------------------------------------
#if USE_TDM
__device__ __forceinline__ void tdm_load_tile(unsigned lds_off,
                                              const _Float16* gptr) {
  const unsigned long long ga = (unsigned long long)(const void*)gptr;
  u32x4 g0;
  g0[0] = 1u;                                            // count=1 (valid), user mode
  g0[1] = lds_off;                                       // lds_addr (bytes)
  g0[2] = (unsigned)ga;                                  // global_addr[31:0]
  g0[3] = (unsigned)((ga >> 32) & 0x01FFFFFFull)         // global_addr[56:32]
          | (2u << 30);                                  // type = 2 ("image")
  i32x8 g1;
  g1[0] = (1 << 16)      // data_size = 1 -> 2 bytes
        | (1 << 20)      // pad_enable
        | (4 << 22)      // pad_interval code 4 -> every 32 dwords (128B)
        | (3 << 25);     // pad_amount  code 3 -> 4 dwords (16B)
  g1[1] = (int)((unsigned)D_DIM << 16);                  // tensor_dim0[15:0] in [31:16]
  g1[2] = (int)(((unsigned)D_DIM >> 16) |
                ((unsigned)N_DIM << 16));                // dim0 hi16 | tensor_dim1 lo16
  g1[3] = (int)(((unsigned)N_DIM >> 16) |
                ((unsigned)KC << 16));                   // dim1 hi16 | tile_dim0
  g1[4] = 128;                                           // tile_dim1 = 128, tile_dim2 = 0
  g1[5] = D_DIM;                                         // tensor_dim0_stride[31:0]
  g1[6] = 0;                                             // dim0_stride hi | dim1_stride lo
  g1[7] = 0;
  const i32x4 z4 = {0, 0, 0, 0};
  const i32x8 z8 = {0, 0, 0, 0, 0, 0, 0, 0};
  __builtin_amdgcn_tensor_load_to_lds(g0, g1, z4, z4, z8, 0);
}
#endif

// ---------------------------------------------------------------------------
// Kernel 4: LDS-staged WMMA GEMM, TDM double buffering.
// Block = 256 threads (8 waves, 4x2): block tile 128(M) x 128(N).
// Wave tile 32x64 = 2x4 WMMA tiles; KC=64 per stage -> 16 wmma/wave/stage.
// ---------------------------------------------------------------------------
__global__ __launch_bounds__(256) void lmme_wmma_gemm_kernel(
    const _Float16* __restrict__ ex, const _Float16* __restrict__ eaT,
    const float* __restrict__ xm, const float* __restrict__ am,
    float* __restrict__ out) {
  __shared__ __align__(16) unsigned char smem[2 * BUF_BYTES];

  const int tid  = threadIdx.x;
  const int lane = tid & 31;
  const int w    = tid >> 5;        // wave 0..7
  const int wr   = w & 3;           // wave row (M): 0..3
  const int wc   = w >> 2;          // wave col (N): 0..1
  const int l16  = lane & 15;
  const int hi   = lane >> 4;       // 0 | 1

  const int bm = blockIdx.y * BLOCK_M;
  const int bn = blockIdx.x * BLOCK_N;
  const int wave_m = wr * 32;       // within block
  const int wave_n = wc * 64;

  const _Float16* gA = ex  + (size_t)bm * D_DIM;   // 128 rows of ex
  const _Float16* gB = eaT + (size_t)bn * D_DIM;   // 128 rows of eaT

  const unsigned smem_base = (unsigned)(unsigned long long)(const void*)smem;

  v8f acc[2][4] = {};

  // ---- stage loader -------------------------------------------------------
  auto stage_load = [&](int buf, int k0) {
#if USE_TDM
    if (w == 0)
      tdm_load_tile(smem_base + buf * BUF_BYTES, gA + k0);
    else if (w == 1)
      tdm_load_tile(smem_base + buf * BUF_BYTES + TILE_BYTES, gB + k0);
#else
    unsigned char* As = smem + buf * BUF_BYTES;
    unsigned char* Bs = As + TILE_BYTES;
    // 128 rows x 4 chunks of 16B (first 32 of the KC=64 halves = 64B per row)
    for (int i = tid; i < 128 * 4; i += 256) {
      const int row = i >> 2, c = i & 3;
      *(v8h*)(As + row * ROWB + c * 16) =
          *(const v8h*)(gA + (size_t)row * D_DIM + k0 + c * 8);
      *(v8h*)(Bs + row * ROWB + c * 16) =
          *(const v8h*)(gB + (size_t)row * D_DIM + k0 + c * 8);
    }
#endif
  };

  stage_load(0, 0);

  int buf = 0;
  for (int s = 0; s < D_DIM / KC; ++s) {
#if USE_TDM
    __builtin_amdgcn_s_wait_tensorcnt(0);  // waves 0/1: DMA for `buf` done
#endif
    __syncthreads();                       // publish `buf`; prior compute done

    if (s + 1 < D_DIM / KC) stage_load(buf ^ 1, (s + 1) * KC);

    const unsigned char* As = smem + buf * BUF_BYTES;
    const unsigned char* Bs = As + TILE_BYTES;

#pragma unroll
    for (int j = 0; j < 2; ++j) {          // two 16x16x32 K-steps per stage
      V16U Af[2], Bf[4];
#pragma unroll
      for (int ti = 0; ti < 2; ++ti) {
        const unsigned char* p =
            As + (wave_m + ti * 16 + l16) * ROWB + (j * 32 + hi * 8) * 2;
        Af[ti].h[0] = *(const v8h*)(p);
        Af[ti].h[1] = *(const v8h*)(p + 32);   // K + 16 halves
      }
#pragma unroll
      for (int tj = 0; tj < 4; ++tj) {
        const unsigned char* q =
            Bs + (wave_n + tj * 16 + l16) * ROWB + (j * 32 + hi * 8) * 2;
        Bf[tj].h[0] = *(const v8h*)(q);
        Bf[tj].h[1] = *(const v8h*)(q + 32);   // K + 16 halves
      }
#pragma unroll
      for (int ti = 0; ti < 2; ++ti)
#pragma unroll
        for (int tj = 0; tj < 4; ++tj)
          acc[ti][tj] = __builtin_amdgcn_wmma_f32_16x16x32_f16(
              false, Af[ti].v, false, Bf[tj].v, (short)0, acc[ti][tj],
              false, false);
    }
    buf ^= 1;
  }

  // ---- epilogue: out = log(acc) + xm[m] + am[n] ---------------------------
#pragma unroll
  for (int ti = 0; ti < 2; ++ti) {
#pragma unroll
    for (int tj = 0; tj < 4; ++tj) {
      const int nIdx = bn + wave_n + tj * 16 + l16;
      const float amv = am[nIdx];
      const int mB = bm + wave_m + ti * 16 + hi * 8;
#pragma unroll
      for (int v = 0; v < 8; ++v) {
        const int m = mB + v;
        out[(size_t)m * E_DIM + nIdx] =
            __logf(acc[ti][tj][v]) + xm[m] + amv;
      }
    }
  }
}

// ---------------------------------------------------------------------------
// Launch
// ---------------------------------------------------------------------------
extern "C" void kernel_launch(void* const* d_in, const int* in_sizes, int n_in,
                              void* d_out, int out_size, void* d_ws, size_t ws_size,
                              hipStream_t stream) {
  const float* x = (const float*)d_in[0];   // (N, D) fp32
  const float* A = (const float*)d_in[1];   // (D, E) fp32
  float* out = (float*)d_out;               // (N, E) fp32

  // Workspace: ex (N*D f16) | eaT (E*D f16) | xm (N f32) | am (E f32)
  _Float16* ex  = (_Float16*)d_ws;
  _Float16* eaT = ex + (size_t)N_DIM * D_DIM;
  float*    xm  = (float*)(eaT + (size_t)E_DIM * D_DIM);
  float*    am  = xm + N_DIM;

  rowmax_exp_kernel<<<N_DIM, 256, 0, stream>>>(x, ex, xm);
  colmax_kernel<<<E_DIM / 256, 256, 0, stream>>>(A, am);
  colexp_transpose_kernel<<<dim3(E_DIM / 32, D_DIM / 32), dim3(32, 8), 0, stream>>>(A, am, eaT);
  lmme_wmma_gemm_kernel<<<dim3(E_DIM / BLOCK_N, N_DIM / BLOCK_M), 256, 0, stream>>>(
      ex, eaT, xm, am, out);
}
